// FeedForward_63127429317249
// MI455X (gfx1250) — matrix operand order repaired and checked by reading, not compiled, not run on hardware
//
#include <hip/hip_runtime.h>
#include <hip/hip_bf16.h>

// Problem constants (LLaMA-7B FFN + LoRA, from the reference)
#define D_MODEL 4096
#define H_FFN   11008
#define N_TOK   4096          // B*S
#define TOK_PER_ADP 1024      // (B/A)*S
#define RANK    16
#define EPSV    1e-5f

#define BM 128
#define BN 64
#define BK 32
#define KP (BK + 8)           // padded LDS row stride (keeps 16B alignment: 40*2=80B)

typedef __bf16 bf16_t;
typedef __attribute__((ext_vector_type(16))) __bf16 v16bf;
typedef __attribute__((ext_vector_type(8)))  __bf16 v8bf;
typedef __attribute__((ext_vector_type(8)))  float  v8f;

// pack two fp32 -> packed bf16 pair (lo = first/even k, hi = second/odd k)
__device__ __forceinline__ unsigned pack_bf16(float lo, float hi) {
  union { unsigned u; __bf16 h[2]; } c;
  c.h[0] = (__bf16)lo;
  c.h[1] = (__bf16)hi;
  return c.u;
}

// ---------------------------------------------------------------------------
// WMMA fragment loaders (wave32 layouts per CDNA5 ISA 7.12.2)
// A 16x32 bf16: lanes 0-15 -> row=lane, K{0..7,16..23}; lanes 16-31 -> K{8..15,24..31}
// ---------------------------------------------------------------------------
__device__ __forceinline__ v16bf frag_a(const bf16_t* base, int lane) {
  const int r  = lane & 15;
  const int kb = (lane >> 4) << 3;                 // 0 or 8
  const v8bf* p = (const v8bf*)(base + r * KP + kb);
  v8bf lo = p[0];                                   // K = kb .. kb+7
  v8bf hi = p[2];                                   // K = kb+16 .. kb+23
  v16bf f;
#pragma unroll
  for (int i = 0; i < 8; ++i) { f[i] = lo[i]; f[i + 8] = hi[i]; }
  return f;
}

// B 32x16 bf16 (stored transposed in LDS: [n][k]):
// lanes 0-15 -> col=lane, K{0..7,16..23} ; lanes 16-31 -> col=lane-16, K{8..15,24..31}
__device__ __forceinline__ v16bf frag_b(const bf16_t* base, int lane) {
  const int c  = lane & 15;
  const int kb = (lane >> 4) << 3;                 // 0 or 8
  const v8bf* p = (const v8bf*)(base + c * KP + kb);
  v8bf lo = p[0];
  v8bf hi = p[2];
  v16bf f;
#pragma unroll
  for (int i = 0; i < 8; ++i) { f[i] = lo[i]; f[i + 8] = hi[i]; }
  return f;
}

// ---------------------------------------------------------------------------
// Kernel 1: RMSNorm -> bf16   (one block per token)
// ---------------------------------------------------------------------------
__global__ __launch_bounds__(256) void k_rmsnorm(const float* __restrict__ data,
                                                 const float* __restrict__ w,
                                                 bf16_t* __restrict__ xout) {
  __shared__ float red[256];
  const int tk = blockIdx.x, t = threadIdx.x;
  const float* row = data + (size_t)tk * D_MODEL;

  float loc[16];
  float ss = 0.f;
  const float4* rp = (const float4*)row;
#pragma unroll
  for (int i = 0; i < 4; ++i) {
    float4 q = rp[t * 4 + i];
    loc[4*i+0] = q.x; loc[4*i+1] = q.y; loc[4*i+2] = q.z; loc[4*i+3] = q.w;
    ss += q.x*q.x + q.y*q.y + q.z*q.z + q.w*q.w;
  }
  red[t] = ss;
  __syncthreads();
  for (int o = 128; o > 0; o >>= 1) {
    if (t < o) red[t] += red[t + o];
    __syncthreads();
  }
  const float inv = rsqrtf(red[0] * (1.0f / (float)D_MODEL) + EPSV);

  const float4* wp = (const float4*)w;
  v8bf pk[2];
#pragma unroll
  for (int i = 0; i < 4; ++i) {
    float4 q = wp[t * 4 + i];
    pk[i >> 1][(i & 1) * 4 + 0] = (bf16_t)(loc[4*i+0] * inv * q.x);
    pk[i >> 1][(i & 1) * 4 + 1] = (bf16_t)(loc[4*i+1] * inv * q.y);
    pk[i >> 1][(i & 1) * 4 + 2] = (bf16_t)(loc[4*i+2] * inv * q.z);
    pk[i >> 1][(i & 1) * 4 + 3] = (bf16_t)(loc[4*i+3] * inv * q.w);
  }
  v8bf* op = (v8bf*)(xout + (size_t)tk * D_MODEL + t * 16);
  op[0] = pk[0];
  op[1] = pk[1];
}

// ---------------------------------------------------------------------------
// Kernel 2/4: rank-16 LoRA down-projection  r[tk][:] = x[tk][:] . A[a][:][:]
// Handles one (A3m!=null: two) matrices.  One block per token.
// ---------------------------------------------------------------------------
__device__ __forceinline__ void block_reduce_rank(float* red, float acc[RANK],
                                                  int t, float* outp) {
#pragma unroll
  for (int r = 0; r < RANK; ++r) red[t * RANK + r] = acc[r];
  __syncthreads();
  for (int o = 128; o > 0; o >>= 1) {
    if (t < o) {
#pragma unroll
      for (int r = 0; r < RANK; ++r) red[t * RANK + r] += red[(t + o) * RANK + r];
    }
    __syncthreads();
  }
  if (t < RANK) outp[t] = red[t];
  __syncthreads();
}

__global__ __launch_bounds__(256) void k_lora_down(const bf16_t* __restrict__ x, int K,
                                                   const float* __restrict__ A1m,
                                                   const float* __restrict__ A3m,
                                                   float* __restrict__ r1,
                                                   float* __restrict__ r3) {
  __shared__ float red[256 * RANK];
  const int tk = blockIdx.x, t = threadIdx.x;
  const int a = tk / TOK_PER_ADP;
  const bf16_t* xr = x + (size_t)tk * K;
  const float* A1p = A1m + (size_t)a * K * RANK;
  const float* A3p = A3m ? (A3m + (size_t)a * K * RANK) : nullptr;

  float acc1[RANK], acc3[RANK];
#pragma unroll
  for (int r = 0; r < RANK; ++r) { acc1[r] = 0.f; acc3[r] = 0.f; }

  for (int d = t; d < K; d += 256) {
    const float xv = (float)xr[d];
    const float* p1 = A1p + (size_t)d * RANK;
#pragma unroll
    for (int r = 0; r < RANK; ++r) acc1[r] += xv * p1[r];
    if (A3p) {
      const float* p3 = A3p + (size_t)d * RANK;
#pragma unroll
      for (int r = 0; r < RANK; ++r) acc3[r] += xv * p3[r];
    }
  }
  block_reduce_rank(red, acc1, t, r1 + (size_t)tk * RANK);
  if (A3m) block_reduce_rank(red, acc3, t, r3 + (size_t)tk * RANK);
}

// ---------------------------------------------------------------------------
// Kernel 3: fused gate GEMM
//   h1 = x.W1 + xr1.B1 ; h3 = x.W3 + xr3.B3 ; act = silu(h1)*h3  (bf16 out)
// Block: 256 thr (8 waves), tile 128(M) x 64(N); wave -> 32x32 per matrix.
// ---------------------------------------------------------------------------
__global__ __launch_bounds__(256) void k_gate(const bf16_t* __restrict__ x,
                                              const float* __restrict__ W1,
                                              const float* __restrict__ W3,
                                              const float* __restrict__ B1g,
                                              const float* __restrict__ B3g,
                                              const float* __restrict__ xr1,
                                              const float* __restrict__ xr3,
                                              bf16_t* __restrict__ act) {
  __shared__ bf16_t As[BM][KP];
  __shared__ bf16_t B1s[BN][KP];
  __shared__ bf16_t B3s[BN][KP];
  __shared__ float  L1s[RANK][BN];
  __shared__ float  L3s[RANK][BN];
  __shared__ float  X1s[BM][RANK];
  __shared__ float  X3s[BM][RANK];

  const int t    = threadIdx.x;
  const int lane = t & 31;
  const int wave = t >> 5;
  const int mw   = (wave >> 1) * 32;     // 0,32,64,96
  const int nw   = (wave & 1) * 32;      // 0,32
  const int n0   = blockIdx.x * BN;
  const int m0   = blockIdx.y * BM;
  const int a    = m0 / TOK_PER_ADP;     // 128 | 1024 -> no straddle

  // stage LoRA operands once
  for (int i = t; i < RANK * BN; i += 256) {
    const int r = i >> 6, n = i & 63;
    L1s[r][n] = B1g[((size_t)a * RANK + r) * H_FFN + n0 + n];
    L3s[r][n] = B3g[((size_t)a * RANK + r) * H_FFN + n0 + n];
  }
  for (int i = t; i < BM * RANK; i += 256) {
    const int m = i >> 4, r = i & 15;
    X1s[m][r] = xr1[(size_t)(m0 + m) * RANK + r];
    X3s[m][r] = xr3[(size_t)(m0 + m) * RANK + r];
  }

  v8f acc1[2][2] = {};
  v8f acc3[2][2] = {};

  const int arow  = t >> 1;              // 0..127
  const int ahalf = (t & 1) << 4;        // 0 or 16 (bf16 elements)
  const int wn4   = (t & 15) << 2;       // 0..60  (4 consecutive n per thread)
  const int wk2   = (t >> 4) << 1;       // 0..30  (2 consecutive k per thread)

  for (int k0 = 0; k0 < D_MODEL; k0 += BK) {
    // x tile: bf16 [BM][BK], two 16B stores per thread
    {
      const v8bf* gp = (const v8bf*)(x + (size_t)(m0 + arow) * D_MODEL + k0 + ahalf);
      v8bf q0 = gp[0], q1 = gp[1];
      v8bf* sp = (v8bf*)(&As[arow][ahalf]);
      sp[0] = q0; sp[1] = q1;
    }
    // W1/W3 tiles: fp32 -> bf16, stored transposed [n][k]; two k packed per b32 store
    {
      const float* w1r0 = W1 + (size_t)(k0 + wk2)     * H_FFN + n0 + wn4;
      const float* w1r1 = W1 + (size_t)(k0 + wk2 + 1) * H_FFN + n0 + wn4;
      const float* w3r0 = W3 + (size_t)(k0 + wk2)     * H_FFN + n0 + wn4;
      const float* w3r1 = W3 + (size_t)(k0 + wk2 + 1) * H_FFN + n0 + wn4;
      float4 a0 = *(const float4*)w1r0;
      float4 a1 = *(const float4*)w1r1;
      float4 c0 = *(const float4*)w3r0;
      float4 c1 = *(const float4*)w3r1;
      if (k0 + BK < D_MODEL) {   // prefetch next k-step weight rows (global_prefetch)
        __builtin_prefetch(w1r0 + (size_t)BK * H_FFN, 0, 1);
        __builtin_prefetch(w3r0 + (size_t)BK * H_FFN, 0, 1);
      }
      float a0v[4] = {a0.x, a0.y, a0.z, a0.w};
      float a1v[4] = {a1.x, a1.y, a1.z, a1.w};
      float c0v[4] = {c0.x, c0.y, c0.z, c0.w};
      float c1v[4] = {c1.x, c1.y, c1.z, c1.w};
#pragma unroll
      for (int j = 0; j < 4; ++j) {
        *(unsigned*)&B1s[wn4 + j][wk2] = pack_bf16(a0v[j], a1v[j]);
        *(unsigned*)&B3s[wn4 + j][wk2] = pack_bf16(c0v[j], c1v[j]);
      }
    }
    __syncthreads();

    const v16bf af0 = frag_a(&As[mw][0],      lane);
    const v16bf af1 = frag_a(&As[mw + 16][0], lane);
    const v16bf b10 = frag_b(&B1s[nw][0],      lane);
    const v16bf b11 = frag_b(&B1s[nw + 16][0], lane);
    const v16bf b30 = frag_b(&B3s[nw][0],      lane);
    const v16bf b31 = frag_b(&B3s[nw + 16][0], lane);

    acc1[0][0] = __builtin_amdgcn_wmma_f32_16x16x32_bf16(false, af0, false, b10, (short)0, acc1[0][0], false, false);
    acc1[0][1] = __builtin_amdgcn_wmma_f32_16x16x32_bf16(false, af0, false, b11, (short)0, acc1[0][1], false, false);
    acc1[1][0] = __builtin_amdgcn_wmma_f32_16x16x32_bf16(false, af1, false, b10, (short)0, acc1[1][0], false, false);
    acc1[1][1] = __builtin_amdgcn_wmma_f32_16x16x32_bf16(false, af1, false, b11, (short)0, acc1[1][1], false, false);
    acc3[0][0] = __builtin_amdgcn_wmma_f32_16x16x32_bf16(false, af0, false, b30, (short)0, acc3[0][0], false, false);
    acc3[0][1] = __builtin_amdgcn_wmma_f32_16x16x32_bf16(false, af0, false, b31, (short)0, acc3[0][1], false, false);
    acc3[1][0] = __builtin_amdgcn_wmma_f32_16x16x32_bf16(false, af1, false, b30, (short)0, acc3[1][0], false, false);
    acc3[1][1] = __builtin_amdgcn_wmma_f32_16x16x32_bf16(false, af1, false, b31, (short)0, acc3[1][1], false, false);
    __syncthreads();
  }

  // epilogue: rank-16 LoRA add + silu-gate, bf16 store
  const int rl = lane & 15;
  const int rh = (lane >> 4) << 3;       // 0 or 8
#pragma unroll
  for (int ms = 0; ms < 2; ++ms)
#pragma unroll
    for (int ns = 0; ns < 2; ++ns) {
#pragma unroll
      for (int i = 0; i < 8; ++i) {
        const int mloc = mw + ms * 16 + i + rh;
        const int nloc = nw + ns * 16 + rl;
        float h1 = acc1[ms][ns][i];
        float h3 = acc3[ms][ns][i];
#pragma unroll
        for (int r = 0; r < RANK; ++r) {
          h1 += X1s[mloc][r] * L1s[r][nloc];
          h3 += X3s[mloc][r] * L3s[r][nloc];
        }
        const float sg = h1 / (1.0f + __expf(-h1));
        act[(size_t)(m0 + mloc) * H_FFN + n0 + nloc] = (bf16_t)(sg * h3);
      }
    }
}

// ---------------------------------------------------------------------------
// Kernel 5: output GEMM  out = act.W2 + actr.B2   (fp32 out)
// ---------------------------------------------------------------------------
__global__ __launch_bounds__(256) void k_out(const bf16_t* __restrict__ act,
                                             const float* __restrict__ W2,
                                             const float* __restrict__ B2g,
                                             const float* __restrict__ actr,
                                             float* __restrict__ out) {
  __shared__ bf16_t As[BM][KP];
  __shared__ bf16_t Bs[BN][KP];
  __shared__ float  L2s[RANK][BN];
  __shared__ float  X2s[BM][RANK];

  const int t    = threadIdx.x;
  const int lane = t & 31;
  const int wave = t >> 5;
  const int mw   = (wave >> 1) * 32;
  const int nw   = (wave & 1) * 32;
  const int n0   = blockIdx.x * BN;
  const int m0   = blockIdx.y * BM;
  const int a    = m0 / TOK_PER_ADP;

  for (int i = t; i < RANK * BN; i += 256) {
    const int r = i >> 6, n = i & 63;
    L2s[r][n] = B2g[((size_t)a * RANK + r) * D_MODEL + n0 + n];
  }
  for (int i = t; i < BM * RANK; i += 256) {
    const int m = i >> 4, r = i & 15;
    X2s[m][r] = actr[(size_t)(m0 + m) * RANK + r];
  }

  v8f acc[2][2] = {};

  const int arow  = t >> 1;
  const int ahalf = (t & 1) << 4;
  const int wn4   = (t & 15) << 2;
  const int wk2   = (t >> 4) << 1;

  for (int k0 = 0; k0 < H_FFN; k0 += BK) {
    {
      const v8bf* gp = (const v8bf*)(act + (size_t)(m0 + arow) * H_FFN + k0 + ahalf);
      v8bf q0 = gp[0], q1 = gp[1];
      v8bf* sp = (v8bf*)(&As[arow][ahalf]);
      sp[0] = q0; sp[1] = q1;
    }
    {
      const float* w2r0 = W2 + (size_t)(k0 + wk2)     * D_MODEL + n0 + wn4;
      const float* w2r1 = W2 + (size_t)(k0 + wk2 + 1) * D_MODEL + n0 + wn4;
      float4 a0 = *(const float4*)w2r0;
      float4 a1 = *(const float4*)w2r1;
      if (k0 + BK < H_FFN)
        __builtin_prefetch(w2r0 + (size_t)BK * D_MODEL, 0, 1);
      float a0v[4] = {a0.x, a0.y, a0.z, a0.w};
      float a1v[4] = {a1.x, a1.y, a1.z, a1.w};
#pragma unroll
      for (int j = 0; j < 4; ++j)
        *(unsigned*)&Bs[wn4 + j][wk2] = pack_bf16(a0v[j], a1v[j]);
    }
    __syncthreads();

    const v16bf af0  = frag_a(&As[mw][0],      lane);
    const v16bf af1  = frag_a(&As[mw + 16][0], lane);
    const v16bf bf0  = frag_b(&Bs[nw][0],      lane);
    const v16bf bf1v = frag_b(&Bs[nw + 16][0], lane);

    acc[0][0] = __builtin_amdgcn_wmma_f32_16x16x32_bf16(false, af0, false, bf0,  (short)0, acc[0][0], false, false);
    acc[0][1] = __builtin_amdgcn_wmma_f32_16x16x32_bf16(false, af0, false, bf1v, (short)0, acc[0][1], false, false);
    acc[1][0] = __builtin_amdgcn_wmma_f32_16x16x32_bf16(false, af1, false, bf0,  (short)0, acc[1][0], false, false);
    acc[1][1] = __builtin_amdgcn_wmma_f32_16x16x32_bf16(false, af1, false, bf1v, (short)0, acc[1][1], false, false);
    __syncthreads();
  }

  const int rl = lane & 15;
  const int rh = (lane >> 4) << 3;
#pragma unroll
  for (int ms = 0; ms < 2; ++ms)
#pragma unroll
    for (int ns = 0; ns < 2; ++ns) {
#pragma unroll
      for (int i = 0; i < 8; ++i) {
        const int mloc = mw + ms * 16 + i + rh;
        const int nloc = nw + ns * 16 + rl;
        float v = acc[ms][ns][i];
#pragma unroll
        for (int r = 0; r < RANK; ++r) v += X2s[mloc][r] * L2s[r][nloc];
        out[(size_t)(m0 + mloc) * D_MODEL + n0 + nloc] = v;
      }
    }
}

// ---------------------------------------------------------------------------
// Host launcher
// ---------------------------------------------------------------------------
extern "C" void kernel_launch(void* const* d_in, const int* in_sizes, int n_in,
                              void* d_out, int out_size, void* d_ws, size_t ws_size,
                              hipStream_t stream) {
  (void)in_sizes; (void)n_in; (void)out_size; (void)ws_size;

  const float* data   = (const float*)d_in[0];
  const float* norm_w = (const float*)d_in[1];
  const float* W1     = (const float*)d_in[2];
  const float* W3     = (const float*)d_in[3];
  const float* W2     = (const float*)d_in[4];
  const float* A1     = (const float*)d_in[5];
  const float* B1     = (const float*)d_in[6];
  const float* A3     = (const float*)d_in[7];
  const float* B3     = (const float*)d_in[8];
  const float* A2     = (const float*)d_in[9];
  const float* B2     = (const float*)d_in[10];
  float* out = (float*)d_out;

  // workspace layout (~119 MB):
  char* ws = (char*)d_ws;
  bf16_t* xbf   = (bf16_t*)ws;                              // 4096*4096*2   = 32 MB
  bf16_t* actbf = (bf16_t*)(ws + (size_t)33554432);         // 4096*11008*2  = 86 MB
  float*  xr1   = (float*)(ws + (size_t)123731968);         // 4096*16*4
  float*  xr3   = xr1 + (size_t)N_TOK * RANK;
  float*  actr  = xr3 + (size_t)N_TOK * RANK;

  k_rmsnorm<<<N_TOK, 256, 0, stream>>>(data, norm_w, xbf);
  k_lora_down<<<N_TOK, 256, 0, stream>>>(xbf, D_MODEL, A1, A3, xr1, xr3);

  dim3 gGate(H_FFN / BN, N_TOK / BM);   // 172 x 32
  k_gate<<<gGate, 256, 0, stream>>>(xbf, W1, W3, B1, B3, xr1, xr3, actbf);

  k_lora_down<<<N_TOK, 256, 0, stream>>>(actbf, H_FFN, A2, nullptr, actr, nullptr);

  dim3 gOut(D_MODEL / BN, N_TOK / BM);  // 64 x 32
  k_out<<<gOut, 256, 0, stream>>>(actbf, W2, B2, actr, out);
}
